// LightningIndexer_53944789237840
// MI455X (gfx1250) — matrix operand order repaired and checked by reading, not compiled, not run on hardware
//
#include <hip/hip_runtime.h>
#include <hip/hip_bf16.h>

// ---------------------------------------------------------------------------
// LightningIndexer for MI455X (gfx1250, wave32, WMMA f32_16x16x32_f16)
//
// scores[b,t,s] = sum_h relu( q[b,t,h,:] . k[b,s,:] ) * w[b,t,h]
//   B=2, T=S=4096, D=1024, H=4, d=32
//
// d == 32 == K of v_wmma_f32_16x16x32_f16  -> one WMMA per (16x16 tile, head)
// Output (128 MiB) is the bandwidth floor; all GEMMs go through matrix cores.
// ---------------------------------------------------------------------------

#define BATCH 2
#define SEQ   4096
#define DMODEL 1024
#define HEADS 4
#define DIDX  32
#define QCOLS 128          // HEADS*DIDX
#define WTROWS 176         // 128 q cols + 32 k cols + 16 (w padded from 4)

typedef __attribute__((ext_vector_type(16))) _Float16 v16h;
typedef __attribute__((ext_vector_type(8)))  _Float16 v8h_t;
typedef __attribute__((ext_vector_type(4)))  _Float16 v4h_t;
typedef __attribute__((ext_vector_type(8)))  float    v8f;
typedef __attribute__((ext_vector_type(4)))  float    v4f;

// A-matrix (16x32 f16) lane layout: lane l holds row l%16,
// halves {klo..klo+7} in VGPR0-3 and {klo+16..klo+23} in VGPR4-7, klo=8*(l/16).
static __device__ __forceinline__ v16h load_a16x32(const _Float16* __restrict__ row, int lh) {
    const int klo = lh * 8;
    union { v16h v; v8h_t h[2]; } u;
    u.h[0] = *(const v8h_t*)(row + klo);
    u.h[1] = *(const v8h_t*)(row + klo + 16);
    return u.v;
}

// B-matrix (32x16 f16) lane layout: lane l holds column l%16,
// K values {kb..kb+7} and {kb+16..kb+23}, kb=8*(l/16)  (two b128 loads).
static __device__ __forceinline__ v16h load_b32x16(const _Float16* __restrict__ colrow, int lh) {
    const int kb = lh * 8;
    union { v16h v; v8h_t h[2]; } u;
    u.h[0] = *(const v8h_t*)(colrow + kb);
    u.h[1] = *(const v8h_t*)(colrow + kb + 16);
    return u.v;
}

static __device__ __forceinline__ v8f wmma_f16(v16h a, v16h b, v8f c) {
    return __builtin_amdgcn_wmma_f32_16x16x32_f16(false, a, false, b, (short)0, c, false, false);
}

// ---------------------------------------------------------------------------
// Kernel 1: x [T,B,D] f32  ->  xh [B,T,D] f16   (batch-major, converted)
// grid = B*T blocks, 256 threads, 4 elements/thread
// ---------------------------------------------------------------------------
__global__ __launch_bounds__(256) void li_convert_x(const float* __restrict__ x,
                                                    _Float16* __restrict__ xh) {
    const long row = blockIdx.x;            // row = b*T + t
    const long b = row >> 12;               // T = 4096
    const long t = row & 4095;
    const float* src = x + (t * BATCH + b) * DMODEL;
    _Float16*    dst = xh + row * DMODEL;
    const int d = threadIdx.x * 4;
    v4f v = *(const v4f*)(src + d);
    v4h_t h;
    h.x = (_Float16)v.x; h.y = (_Float16)v.y; h.z = (_Float16)v.z; h.w = (_Float16)v.w;
    *(v4h_t*)(dst + d) = h;
}

// ---------------------------------------------------------------------------
// Kernel 2: build transposed f16 weight matrix Wt[176][1024]
//   rows 0..127   : Wq columns   (Wq is [1024][128])
//   rows 128..159 : Wk columns   (Wk is [1024][32])
//   rows 160..163 : Ww columns   (Ww is [1024][4])
//   rows 164..175 : zero padding (so the w tile is a full 16-col WMMA tile)
// grid = 176 blocks, 256 threads, 4 k/thread
// ---------------------------------------------------------------------------
__global__ __launch_bounds__(256) void li_build_wt(const float* __restrict__ Wq,
                                                   const float* __restrict__ Wk,
                                                   const float* __restrict__ Ww,
                                                   _Float16* __restrict__ Wt) {
    const int j = blockIdx.x;               // 0..175 (block-uniform)
    _Float16* dst = Wt + (long)j * DMODEL;
    for (int ii = 0; ii < 4; ++ii) {
        const int i = threadIdx.x + ii * 256;
        float v;
        if (j < 128)      v = Wq[(long)i * QCOLS + j];
        else if (j < 160) v = Wk[(long)i * DIDX + (j - 128)];
        else if (j < 164) v = Ww[(long)i * HEADS + (j - 160)];
        else              v = 0.0f;
        dst[i] = (_Float16)v;
    }
}

// ---------------------------------------------------------------------------
// Kernel 3: projection GEMM via WMMA.
//   [16 t-rows] x [16 cols] tile, K-loop over 1024 in steps of 32.
//   11 col-tiles: 8 -> qh (f16), 2 -> kh (f16), 1 -> wf (f32, 4 valid cols).
// One wave per (b, t-tile, col-tile): 2*256*11 = 5632 waves = 704 blocks x 8.
// ---------------------------------------------------------------------------
__global__ __launch_bounds__(256) void li_proj_wmma(const _Float16* __restrict__ xh,
                                                    const _Float16* __restrict__ Wt,
                                                    const float* __restrict__ bq,
                                                    const float* __restrict__ bk,
                                                    const float* __restrict__ bw,
                                                    _Float16* __restrict__ qh,
                                                    _Float16* __restrict__ kh,
                                                    float* __restrict__ wf) {
    const int wave = threadIdx.x >> 5;
    const int lane = threadIdx.x & 31;
    const int task = blockIdx.x * 8 + wave;        // 0..5631
    const int ct    = task % 11;                   // col tile
    const int ttile = (task / 11) & 255;           // t tile within batch
    const int b     = task / (11 * 256);
    const int lm = lane & 15;
    const int lh = lane >> 4;

    const _Float16* aRow = xh + ((long)b * SEQ + ttile * 16 + lm) * DMODEL;
    const _Float16* bRow = Wt + (long)(ct * 16 + lm) * DMODEL;

    v8f c = {};
    #pragma unroll 4
    for (int kk = 0; kk < DMODEL; kk += 32) {
        v16h a  = load_a16x32(aRow + kk, lh);
        v16h bm = load_b32x16(bRow + kk, lh);
        c = wmma_f16(a, bm, c);
    }

    // Epilogue (divergence allowed now; WMMA loop is done with full EXEC).
    const int n = lm;
    float bias;
    if (ct < 8)       bias = bq[ct * 16 + n];
    else if (ct < 10) bias = bk[(ct - 8) * 16 + n];
    else              bias = (n < 4) ? bw[n] : 0.0f;

    #pragma unroll
    for (int r = 0; r < 8; ++r) {
        const int m = r + 8 * lh;
        const long trow = (long)b * SEQ + ttile * 16 + m;
        const float v = c[r] + bias;
        if (ct < 8)       qh[trow * QCOLS + ct * 16 + n] = (_Float16)v;
        else if (ct < 10) kh[trow * DIDX + (ct - 8) * 16 + n] = (_Float16)v;
        else if (n < 4)   wf[trow * HEADS + n] = v;
    }
}

// ---------------------------------------------------------------------------
// Kernel 4: indexer core.
//   Per wave: one 16-row t-tile (A for all 4 heads held in registers) and a
//   chunk of 32 s-tiles. Per s-tile: load k tile (B, shared by heads), 4 WMMAs,
//   relu + head-weighted combine on v8f accumulators, 8 NT f32 stores/lane.
// 2*256 t-tiles * 8 s-chunks = 4096 waves = 512 blocks x 8 (all EXEC full).
// ---------------------------------------------------------------------------
__global__ __launch_bounds__(256) void li_indexer_wmma(const _Float16* __restrict__ qh,
                                                       const _Float16* __restrict__ kh,
                                                       const float* __restrict__ wf,
                                                       float* __restrict__ out) {
    const int wave = threadIdx.x >> 5;
    const int lane = threadIdx.x & 31;
    const int task = blockIdx.x * 8 + wave;        // 0..4095
    const int schunk = task & 7;
    const int ttile  = (task >> 3) & 255;
    const int b      = task >> 11;
    const int lm = lane & 15;
    const int lh = lane >> 4;

    const long tbase = (long)b * SEQ + ttile * 16;

    // A matrices, one per head (q rows t0..t0+15, 32 halves each head).
    const _Float16* aRow = qh + (tbase + lm) * QCOLS;
    v16h A0 = load_a16x32(aRow + 0 * DIDX, lh);
    v16h A1 = load_a16x32(aRow + 1 * DIDX, lh);
    v16h A2 = load_a16x32(aRow + 2 * DIDX, lh);
    v16h A3 = load_a16x32(aRow + 3 * DIDX, lh);

    // Head weights for this lane's output rows m = r + 8*lh.
    v4f wv[8];
    #pragma unroll
    for (int r = 0; r < 8; ++r)
        wv[r] = *(const v4f*)(wf + (tbase + r + 8 * lh) * HEADS);

    const _Float16* kbat = kh + (long)b * SEQ * DIDX;
    float* orow = out + tbase * SEQ;

    for (int st = schunk * 32; st < schunk * 32 + 32; ++st) {
        const int s0 = st * 16;
        v16h Bm = load_b32x16(kbat + (long)(s0 + lm) * DIDX, lh);
        v8f d0 = {}, d1 = {}, d2 = {}, d3 = {};
        d0 = wmma_f16(A0, Bm, d0);
        d1 = wmma_f16(A1, Bm, d1);
        d2 = wmma_f16(A2, Bm, d2);
        d3 = wmma_f16(A3, Bm, d3);
        #pragma unroll
        for (int r = 0; r < 8; ++r) {
            const float v = fmaxf(d0[r], 0.0f) * wv[r].x
                          + fmaxf(d1[r], 0.0f) * wv[r].y
                          + fmaxf(d2[r], 0.0f) * wv[r].z
                          + fmaxf(d3[r], 0.0f) * wv[r].w;
            const long o = (long)(r + 8 * lh) * SEQ + s0 + lm;
            __builtin_nontemporal_store(v, orow + o);
        }
    }
}

// ---------------------------------------------------------------------------
// Workspace layout (bytes), ~19.9 MiB total:
//   xh : B*T*1024 f16 = 16 MiB          @ 0
//   qh : B*T*128  f16 = 2 MiB           @ 16777216
//   kh : B*T*32   f16 = 512 KiB         @ 18874368
//   wf : B*T*4    f32 = 128 KiB         @ 19398656
//   Wt : 176*1024 f16 = 352 KiB         @ 19529728
// ---------------------------------------------------------------------------
extern "C" void kernel_launch(void* const* d_in, const int* in_sizes, int n_in,
                              void* d_out, int out_size, void* d_ws, size_t ws_size,
                              hipStream_t stream) {
    const float* x  = (const float*)d_in[0];
    const float* Wq = (const float*)d_in[1];
    const float* bq = (const float*)d_in[2];
    const float* Wk = (const float*)d_in[3];
    const float* bk = (const float*)d_in[4];
    const float* Ww = (const float*)d_in[5];
    const float* bw = (const float*)d_in[6];
    float* out = (float*)d_out;

    char* ws = (char*)d_ws;
    _Float16* xh = (_Float16*)(ws + 0);
    _Float16* qh = (_Float16*)(ws + 16777216);
    _Float16* kh = (_Float16*)(ws + 18874368);
    float*    wf = (float*)   (ws + 19398656);
    _Float16* Wt = (_Float16*)(ws + 19529728);

    li_convert_x  <<<BATCH * SEQ, 256, 0, stream>>>(x, xh);
    li_build_wt   <<<WTROWS, 256, 0, stream>>>(Wq, Wk, Ww, Wt);
    li_proj_wmma  <<<704, 256, 0, stream>>>(xh, Wt, bq, bk, bw, qh, kh, wf);
    li_indexer_wmma<<<512, 256, 0, stream>>>(qh, kh, wf, out);
}
